// RoutedSelfAttention_37220186587175
// MI455X (gfx1250) — matrix operand order repaired and checked by reading, not compiled, not run on hardware
//
#include <hip/hip_runtime.h>

// ---------------- problem constants ----------------
constexpr int kB  = 32;    // batch
constexpr int kL  = 512;   // sequence length
constexpr int kD  = 1024;  // model dim
constexpr int kE  = 8;     // experts
constexpr int kH  = 16;    // heads
constexpr int kHD = 64;    // head dim

// ---------------- vector types ----------------
typedef __attribute__((ext_vector_type(16))) __bf16        v16bf;
typedef __attribute__((ext_vector_type(8)))  float         v8f;
typedef __attribute__((ext_vector_type(8)))  unsigned short u16x8;
typedef __attribute__((ext_vector_type(4)))  unsigned short u16x4;
typedef __attribute__((ext_vector_type(4)))  float          f32x4;
typedef __attribute__((ext_vector_type(4)))  int            v4i;

union FragCvt {
    struct { u16x8 lo; u16x8 hi; } s;
    v16bf v;
};

static __device__ __forceinline__ v16bf make_frag(u16x8 lo, u16x8 hi) {
    FragCvt f; f.s.lo = lo; f.s.hi = hi; return f.v;
}

static __device__ __forceinline__ v8f v8f_zero() {
    v8f z = {0.f,0.f,0.f,0.f,0.f,0.f,0.f,0.f}; return z;
}

static __device__ __forceinline__ unsigned short f32_to_bf16(float f) {
    unsigned int u = __float_as_uint(f);
    unsigned int r = u + 0x7FFFu + ((u >> 16) & 1u);   // round-to-nearest-even
    return (unsigned short)(r >> 16);
}

// WMMA: D = A(16x32 bf16) * B(32x16 bf16) + C(16x16 f32)
static __device__ __forceinline__ v8f wmma_bf16(v16bf a, v16bf b, v8f c) {
    return __builtin_amdgcn_wmma_f32_16x16x32_bf16(false, a, false, b,
                                                   (short)0, c, false, false);
}

// -------- async global->LDS copy (16 bytes), CDNA5 ASYNCcnt path --------
#if defined(__has_builtin)
#if __has_builtin(__builtin_amdgcn_global_load_async_to_lds_b128)
#define HAVE_ASYNC_BUILTIN 1
#endif
#if __has_builtin(__builtin_amdgcn_s_wait_asynccnt)
#define HAVE_ASYNCWAIT_BUILTIN 1
#endif
#endif

// builtin expects pointers to 4xint vectors (per hipcc diagnostic)
typedef __attribute__((address_space(1))) v4i gv4i;
typedef __attribute__((address_space(3))) v4i lv4i;

static __device__ __forceinline__ void async_copy16(const void* g, void* l) {
#ifdef HAVE_ASYNC_BUILTIN
    gv4i* gp = (gv4i*)(unsigned long long)(uintptr_t)g;
    lv4i* lp = (lv4i*)(unsigned int)(uintptr_t)l;
    __builtin_amdgcn_global_load_async_to_lds_b128(gp, lp, 0, 0);
#else
    asm volatile("global_load_async_to_lds_b128 %0, %1, off"
                 :: "v"((unsigned int)(uintptr_t)l),
                    "v"((unsigned long long)(uintptr_t)g)
                 : "memory");
#endif
}

static __device__ __forceinline__ void wait_async0() {
#ifdef HAVE_ASYNCWAIT_BUILTIN
    __builtin_amdgcn_s_wait_asynccnt(0);
#else
    asm volatile("s_wait_asynccnt 0" ::: "memory");
#endif
}

// A-fragment 16x32 (MxK) from an [m][k] layout (LDS or global), stride in elems.
static __device__ __forceinline__ v16bf load_a16x32(const unsigned short* p,
                                                    int stride, int lane) {
    int m  = lane & 15;
    int h8 = (lane >> 4) << 3;
    const unsigned short* row = p + m * stride + h8;
    u16x8 lo = *(const u16x8*)(row);
    u16x8 hi = *(const u16x8*)(row + 16);
    return make_frag(lo, hi);
}

// B-fragment 32x16 (KxN) from an [n][k] layout, stride in elems.
static __device__ __forceinline__ v16bf load_b32x16(const unsigned short* p,
                                                    int stride, int lane) {
    int n   = lane & 15;
    int klo = (lane >> 4) << 3;
    const unsigned short* row = p + n * stride + klo;
    u16x8 a = *(const u16x8*)(row);
    u16x8 b = *(const u16x8*)(row + 16);
    return make_frag(a, b);
}

// =====================================================================
// Kernel 0: one-shot f32 -> bf16 convert of x (feeds async A staging)
// =====================================================================
__global__ __launch_bounds__(256) void xcvt_kernel(const float* __restrict__ x,
                                                   unsigned short* __restrict__ xbf) {
    size_t i = ((size_t)blockIdx.x * 256 + threadIdx.x) * 8;
    f32x4 a = *(const f32x4*)(x + i);
    f32x4 b = *(const f32x4*)(x + i + 4);
    u16x8 u;
    u[0] = f32_to_bf16(a.x); u[1] = f32_to_bf16(a.y);
    u[2] = f32_to_bf16(a.z); u[3] = f32_to_bf16(a.w);
    u[4] = f32_to_bf16(b.x); u[5] = f32_to_bf16(b.y);
    u[6] = f32_to_bf16(b.z); u[7] = f32_to_bf16(b.w);
    *(u16x8*)(xbf + i) = u;
}

// =====================================================================
// Kernel 1: router — mean pool, logits, softmax probs, argmax route
// =====================================================================
__global__ __launch_bounds__(256) void router_kernel(
        const float* __restrict__ x, const float* __restrict__ rw,
        const float* __restrict__ rb, float* __restrict__ probs_out,
        int* __restrict__ routes) {
    __shared__ float part[8][kD];
    __shared__ float meanv[kD];
    __shared__ float logits[kE];

    int b = blockIdx.x;
    int t = threadIdx.x, w = t >> 5, lane = t & 31;
    const float* xb = x + (size_t)b * kL * kD;

    float acc[32];
#pragma unroll
    for (int i = 0; i < 32; ++i) acc[i] = 0.f;
    for (int l = w * 64; l < w * 64 + 64; ++l) {
        const float* row = xb + (size_t)l * kD;
#pragma unroll
        for (int i = 0; i < 32; ++i) acc[i] += row[lane + 32 * i];
    }
#pragma unroll
    for (int i = 0; i < 32; ++i) part[w][lane + 32 * i] = acc[i];
    __syncthreads();

#pragma unroll
    for (int i = 0; i < 4; ++i) {
        int d = t * 4 + i;
        float s = 0.f;
#pragma unroll
        for (int ww = 0; ww < 8; ++ww) s += part[ww][d];
        meanv[d] = s * (1.0f / (float)kL);
    }
    __syncthreads();

    {   // 8 waves -> 8 experts
        int e = w;
        float s = 0.f;
        for (int d = lane; d < kD; d += 32) s += meanv[d] * rw[d * kE + e];
#pragma unroll
        for (int off = 16; off; off >>= 1) s += __shfl_xor(s, off, 32);
        if (lane == 0) logits[e] = s + rb[e];
    }
    __syncthreads();

    if (t == 0) {
        float mx = -1e30f;
#pragma unroll
        for (int e = 0; e < kE; ++e) mx = fmaxf(mx, logits[e]);
        float p[kE], se = 0.f;
#pragma unroll
        for (int e = 0; e < kE; ++e) { p[e] = __expf(logits[e] - mx); se += p[e]; }
        int best = 0; float bl = logits[0];
#pragma unroll
        for (int e = 1; e < kE; ++e) if (logits[e] > bl) { bl = logits[e]; best = e; }
        float inv = 1.0f / se;
#pragma unroll
        for (int e = 0; e < kE; ++e) probs_out[b * kE + e] = p[e] * inv;
        routes[b] = best;
    }
}

// =====================================================================
// Kernel 2: QKV projection GEMM (bf16 WMMA, async A staging, fused RoPE)
// block: 256 thr (8 waves, 4x2), tile 128x128x32
// Q,K -> [B,H,L,hd] bf16 (RoPE applied); V -> [B,H,hd,L] bf16 (transposed)
// =====================================================================
constexpr int BM = 128, BN = 128, BK = 32, LDP = 40;

__global__ __launch_bounds__(256) void qkv_kernel(
        const unsigned short* __restrict__ xbf,
        const float* __restrict__ qw, const float* __restrict__ kw,
        const float* __restrict__ vw, const int* __restrict__ routes,
        unsigned short* __restrict__ qbf, unsigned short* __restrict__ kbf,
        unsigned short* __restrict__ vbf) {
    __shared__ unsigned short As[BM * LDP];   // [m][k]
    __shared__ unsigned short Bs[BN * LDP];   // [n][k]  (transposed)

    const int tiles = (kL / BM) * (kD / BN);  // 32
    int idx  = blockIdx.x;
    int b    = idx / (3 * tiles);
    int mat  = (idx / tiles) % 3;
    int tile = idx % tiles;
    int tm   = tile / (kD / BN);
    int tn   = tile % (kD / BN);

    int route = routes[b];
    const float* W = (mat == 0 ? qw : (mat == 1 ? kw : vw)) + (size_t)route * kD * kD;
    const unsigned short* A = xbf + (size_t)b * kL * kD;

    int t = threadIdx.x, wid = t >> 5, lane = t & 31;
    int wm = wid & 3, wn = wid >> 2;          // wave tile: 32 x 64

    v8f acc[2][4];
#pragma unroll
    for (int mf = 0; mf < 2; ++mf)
#pragma unroll
        for (int nf = 0; nf < 4; ++nf) acc[mf][nf] = v8f_zero();

    for (int k0 = 0; k0 < kD; k0 += BK) {
        // stage A tile 128x32 bf16: async global->LDS (16B chunks)
#pragma unroll
        for (int c = 0; c < 2; ++c) {
            int lin = c * 256 + t;
            int row = lin >> 2, kq = (lin & 3) << 3;
            async_copy16(A + (size_t)(tm * BM + row) * kD + k0 + kq,
                         &As[row * LDP + kq]);
        }
        // stage B tile 32x128 f32 -> bf16 LDS transposed [n][k]
#pragma unroll
        for (int c = 0; c < 4; ++c) {
            int lin = c * 256 + t;
            int kr = lin >> 5, nq = (lin & 31) << 2;
            f32x4 v = *(const f32x4*)(W + (size_t)(k0 + kr) * kD + tn * BN + nq);
            Bs[(nq + 0) * LDP + kr] = f32_to_bf16(v.x);
            Bs[(nq + 1) * LDP + kr] = f32_to_bf16(v.y);
            Bs[(nq + 2) * LDP + kr] = f32_to_bf16(v.z);
            Bs[(nq + 3) * LDP + kr] = f32_to_bf16(v.w);
        }
        if (k0 + BK < kD)   // hint next W tile toward L2
            __builtin_prefetch(W + (size_t)(k0 + BK + (t >> 3)) * kD + tn * BN, 0, 1);
        wait_async0();
        __syncthreads();

        v16bf af[2], bf[4];
#pragma unroll
        for (int mf = 0; mf < 2; ++mf)
            af[mf] = load_a16x32(&As[(wm * 32 + mf * 16) * LDP], LDP, lane);
#pragma unroll
        for (int nf = 0; nf < 4; ++nf)
            bf[nf] = load_b32x16(&Bs[(wn * 64 + nf * 16) * LDP], LDP, lane);
#pragma unroll
        for (int mf = 0; mf < 2; ++mf)
#pragma unroll
            for (int nf = 0; nf < 4; ++nf)
                acc[mf][nf] = wmma_bf16(af[mf], bf[nf], acc[mf][nf]);
        __syncthreads();
    }

    // epilogue: RoPE (Q,K) + bf16 store with head-major layouts
    int hw = lane >> 4, nl = lane & 15;
#pragma unroll
    for (int mf = 0; mf < 2; ++mf)
#pragma unroll
    for (int nf = 0; nf < 4; ++nf)
#pragma unroll
    for (int rr = 0; rr < 8; ++rr) {
        float v = acc[mf][nf][rr];
        int lrow = tm * BM + wm * 32 + mf * 16 + rr + 8 * hw;   // seq pos
        int c    = tn * BN + wn * 64 + nf * 16 + nl;            // out column
        float res = v;
        if (mat < 2) {  // interleaved RoPE: partner column c^1 sits in lane^1
            float p = __shfl_xor(v, 1, 32);
            int d  = c & (kHD - 1);
            int jj = d >> 1;
            float inv_freq = __expf((float)jj * (-2.0f * 9.210340371976184f / (float)kHD));
            float ang = (float)lrow * inv_freq;
            float sn, cs; __sincosf(ang, &sn, &cs);
            res = (d & 1) ? fmaf(v, cs, p * sn) : fmaf(v, cs, -p * sn);
        }
        unsigned short ub = f32_to_bf16(res);
        int hh = c >> 6, dd = c & (kHD - 1);
        if (mat == 2)        // V transposed: [B,H,hd,L]
            vbf[(((size_t)b * kH + hh) * kHD + dd) * kL + lrow] = ub;
        else if (mat == 0)   // Q: [B,H,L,hd]
            qbf[(((size_t)b * kH + hh) * kL + lrow) * kHD + dd] = ub;
        else                 // K: [B,H,L,hd]
            kbf[(((size_t)b * kH + hh) * kL + lrow) * kHD + dd] = ub;
    }
}

// =====================================================================
// Kernel 3: causal flash attention per (b,h).
// block: 128 thr (4 waves); wave owns 32 query rows; 32-key steps.
// =====================================================================
__global__ __launch_bounds__(128) void attn_kernel(
        const unsigned short* __restrict__ qbf,
        const unsigned short* __restrict__ kbf,
        const unsigned short* __restrict__ vbf,
        unsigned short* __restrict__ ctx) {
    __shared__ unsigned short Ps[4][32 * LDP];   // per-wave P staging

    int idx = blockIdx.x;
    int tq = idx & 3, h = (idx >> 2) & (kH - 1), b = idx >> 6;
    int t = threadIdx.x, w = t >> 5, lane = t & 31;
    int hw = lane >> 4, nl = lane & 15;
    int q0 = tq * 128 + w * 32;

    const unsigned short* qb = qbf + ((size_t)b * kH + h) * kL * kHD;
    const unsigned short* kb = kbf + ((size_t)b * kH + h) * kL * kHD;
    const unsigned short* vb = vbf + ((size_t)b * kH + h) * kHD * kL;

    v16bf qa[2][2];
#pragma unroll
    for (int mf = 0; mf < 2; ++mf)
#pragma unroll
        for (int kd = 0; kd < 2; ++kd)
            qa[mf][kd] = load_a16x32(qb + (size_t)(q0 + mf * 16) * kHD + kd * 32,
                                     kHD, lane);

    v8f acc[2][4];
#pragma unroll
    for (int mf = 0; mf < 2; ++mf)
#pragma unroll
        for (int nv = 0; nv < 4; ++nv) acc[mf][nv] = v8f_zero();

    float mrow[16], lrow[16];
#pragma unroll
    for (int i = 0; i < 16; ++i) { mrow[i] = -1e30f; lrow[i] = 0.f; }

    const int jmax = tq * 4 + 3;       // uniform over block (barrier safety)
    const int jme  = tq * 4 + w;       // this wave's causal limit
    for (int j = 0; j <= jmax; ++j) {
        bool act = (j <= jme);
        if (act) {
            v16bf kf[2][2];
#pragma unroll
            for (int nf = 0; nf < 2; ++nf)
#pragma unroll
                for (int kd = 0; kd < 2; ++kd)
                    kf[nf][kd] = load_b32x16(
                        kb + (size_t)(j * 32 + nf * 16) * kHD + kd * 32, kHD, lane);

            v8f s[2][2];
#pragma unroll
            for (int mf = 0; mf < 2; ++mf)
#pragma unroll
                for (int nf = 0; nf < 2; ++nf) {
                    v8f sv = wmma_bf16(qa[mf][0], kf[nf][0], v8f_zero());
                    s[mf][nf] = wmma_bf16(qa[mf][1], kf[nf][1], sv);
                }

            bool diag = (j == jme);
#pragma unroll
            for (int mf = 0; mf < 2; ++mf)
#pragma unroll
                for (int nf = 0; nf < 2; ++nf)
#pragma unroll
                    for (int rr = 0; rr < 8; ++rr) {
                        float sv = s[mf][nf][rr] * 0.125f;   // hd^-0.5
                        if (diag) {
                            int key = j * 32 + nf * 16 + nl;
                            int qr  = q0 + mf * 16 + rr + 8 * hw;
                            if (key > qr) sv = -1e9f;
                        }
                        s[mf][nf][rr] = sv;
                    }

            // online softmax per row (rows live in 16-lane half-waves)
#pragma unroll
            for (int mf = 0; mf < 2; ++mf)
#pragma unroll
                for (int rr = 0; rr < 8; ++rr) {
                    float rm = fmaxf(s[mf][0][rr], s[mf][1][rr]);
#pragma unroll
                    for (int off = 1; off < 16; off <<= 1)
                        rm = fmaxf(rm, __shfl_xor(rm, off, 32));
                    int ri = mf * 8 + rr;
                    float mnew = fmaxf(mrow[ri], rm);
                    float corr = __expf(mrow[ri] - mnew);
                    mrow[ri] = mnew;
                    float e0 = __expf(s[mf][0][rr] - mnew);
                    float e1 = __expf(s[mf][1][rr] - mnew);
                    s[mf][0][rr] = e0; s[mf][1][rr] = e1;
                    float rs = e0 + e1;
#pragma unroll
                    for (int off = 1; off < 16; off <<= 1)
                        rs += __shfl_xor(rs, off, 32);
                    lrow[ri] = lrow[ri] * corr + rs;
#pragma unroll
                    for (int nv = 0; nv < 4; ++nv) acc[mf][nv][rr] *= corr;
                }

            // stage P (C-layout -> [m][k] bf16 in LDS)
#pragma unroll
            for (int mf = 0; mf < 2; ++mf)
#pragma unroll
                for (int nf = 0; nf < 2; ++nf)
#pragma unroll
                    for (int rr = 0; rr < 8; ++rr)
                        Ps[w][(mf * 16 + rr + 8 * hw) * LDP + nf * 16 + nl] =
                            f32_to_bf16(s[mf][nf][rr]);
        }
        __syncthreads();
        if (act) {
            v16bf pa[2];
            pa[0] = load_a16x32(&Ps[w][0],        LDP, lane);
            pa[1] = load_a16x32(&Ps[w][16 * LDP], LDP, lane);
#pragma unroll
            for (int nv = 0; nv < 4; ++nv) {
                v16bf vf = load_b32x16(vb + (size_t)(nv * 16) * kL + j * 32, kL, lane);
#pragma unroll
                for (int mf = 0; mf < 2; ++mf)
                    acc[mf][nv] = wmma_bf16(pa[mf], vf, acc[mf][nv]);
            }
        }
        __syncthreads();
    }

    float invl[16];
#pragma unroll
    for (int i = 0; i < 16; ++i) invl[i] = 1.0f / lrow[i];
#pragma unroll
    for (int mf = 0; mf < 2; ++mf)
#pragma unroll
    for (int nv = 0; nv < 4; ++nv)
#pragma unroll
    for (int rr = 0; rr < 8; ++rr) {
        float o = acc[mf][nv][rr] * invl[mf * 8 + rr];
        int lq = q0 + mf * 16 + rr + 8 * hw;
        int c  = h * kHD + nv * 16 + nl;
        ctx[((size_t)b * kL + lq) * kD + c] = f32_to_bf16(o);
    }
}

// =====================================================================
// Kernel 4: output projection, bf16 ctx x routed out_projs -> f32 d_out
// (A side staged with async global->LDS copies)
// =====================================================================
__global__ __launch_bounds__(256) void outproj_kernel(
        const unsigned short* __restrict__ ctx, const float* __restrict__ ow,
        const int* __restrict__ routes, float* __restrict__ out) {
    __shared__ unsigned short As[BM * LDP];
    __shared__ unsigned short Bs[BN * LDP];

    const int tiles = (kL / BM) * (kD / BN);  // 32
    int idx = blockIdx.x;
    int b = idx / tiles, tile = idx % tiles;
    int tm = tile / (kD / BN), tn = tile % (kD / BN);

    int route = routes[b];
    const unsigned short* A = ctx + (size_t)b * kL * kD;
    const float* W = ow + (size_t)route * kD * kD;

    int t = threadIdx.x, wid = t >> 5, lane = t & 31;
    int wm = wid & 3, wn = wid >> 2;

    v8f acc[2][4];
#pragma unroll
    for (int mf = 0; mf < 2; ++mf)
#pragma unroll
        for (int nf = 0; nf < 4; ++nf) acc[mf][nf] = v8f_zero();

    for (int k0 = 0; k0 < kD; k0 += BK) {
        // A already bf16: async copies into LDS [m][k]
#pragma unroll
        for (int c = 0; c < 2; ++c) {
            int lin = c * 256 + t;
            int row = lin >> 2, kq = (lin & 3) << 3;
            async_copy16(A + (size_t)(tm * BM + row) * kD + k0 + kq,
                         &As[row * LDP + kq]);
        }
#pragma unroll
        for (int c = 0; c < 4; ++c) {
            int lin = c * 256 + t;
            int kr = lin >> 5, nq = (lin & 31) << 2;
            f32x4 v = *(const f32x4*)(W + (size_t)(k0 + kr) * kD + tn * BN + nq);
            Bs[(nq + 0) * LDP + kr] = f32_to_bf16(v.x);
            Bs[(nq + 1) * LDP + kr] = f32_to_bf16(v.y);
            Bs[(nq + 2) * LDP + kr] = f32_to_bf16(v.z);
            Bs[(nq + 3) * LDP + kr] = f32_to_bf16(v.w);
        }
        if (k0 + BK < kD)
            __builtin_prefetch(W + (size_t)(k0 + BK + (t >> 3)) * kD + tn * BN, 0, 1);
        wait_async0();
        __syncthreads();

        v16bf af[2], bf[4];
#pragma unroll
        for (int mf = 0; mf < 2; ++mf)
            af[mf] = load_a16x32(&As[(wm * 32 + mf * 16) * LDP], LDP, lane);
#pragma unroll
        for (int nf = 0; nf < 4; ++nf)
            bf[nf] = load_b32x16(&Bs[(wn * 64 + nf * 16) * LDP], LDP, lane);
#pragma unroll
        for (int mf = 0; mf < 2; ++mf)
#pragma unroll
            for (int nf = 0; nf < 4; ++nf)
                acc[mf][nf] = wmma_bf16(af[mf], bf[nf], acc[mf][nf]);
        __syncthreads();
    }

    int hw = lane >> 4, nl = lane & 15;
#pragma unroll
    for (int mf = 0; mf < 2; ++mf)
#pragma unroll
    for (int nf = 0; nf < 4; ++nf)
#pragma unroll
    for (int rr = 0; rr < 8; ++rr) {
        int lrow = tm * BM + wm * 32 + mf * 16 + rr + 8 * hw;
        int c    = tn * BN + wn * 64 + nf * 16 + nl;
        out[((size_t)b * kL + lrow) * kD + c] = acc[mf][nf][rr];
    }
}

// =====================================================================
// Host launch
// =====================================================================
extern "C" void kernel_launch(void* const* d_in, const int* in_sizes, int n_in,
                              void* d_out, int out_size, void* d_ws, size_t ws_size,
                              hipStream_t stream) {
    const float* x  = (const float*)d_in[0];
    const float* qw = (const float*)d_in[1];
    const float* kw = (const float*)d_in[2];
    const float* vw = (const float*)d_in[3];
    const float* ow = (const float*)d_in[4];
    const float* rw = (const float*)d_in[5];
    const float* rb = (const float*)d_in[6];

    float* out   = (float*)d_out;
    float* probs = out + (size_t)kB * kL * kD;   // second tuple output

    // workspace layout: routes | x_bf | q_bf | k_bf | v_bf | ctx_bf (~160 MB)
    char* ws = (char*)d_ws;
    int* routes = (int*)ws;
    const size_t elems = (size_t)kB * kL * kD;
    unsigned short* xbf = (unsigned short*)(ws + 256);
    unsigned short* qbf = xbf + elems;
    unsigned short* kbf = qbf + elems;
    unsigned short* vbf = kbf + elems;
    unsigned short* ctx = vbf + elems;

    xcvt_kernel   <<<(int)(elems / (256 * 8)), 256, 0, stream>>>(x, xbf);
    router_kernel <<<kB,            256, 0, stream>>>(x, rw, rb, probs, routes);
    qkv_kernel    <<<kB * 3 * 32,   256, 0, stream>>>(xbf, qw, kw, vw, routes,
                                                      qbf, kbf, vbf);
    attn_kernel   <<<kB * kH * 4,   128, 0, stream>>>(qbf, kbf, vbf, ctx);
    outproj_kernel<<<kB * 32,       256, 0, stream>>>(ctx, ow, routes, out);
}
